// RNModel_36584531427738
// MI455X (gfx1250) — matrix operand order repaired and checked by reading, not compiled, not run on hardware
//
#include <hip/hip_runtime.h>
#include <stdint.h>

typedef __attribute__((ext_vector_type(16))) __bf16 v16bf;
typedef __attribute__((ext_vector_type(8)))  float  v8f;
typedef __attribute__((ext_vector_type(4)))  int    v4i;

#define BB  4
#define NN  512
#define DIN 32
#define LL  64
#define HH  128

// ---------------------------------------------------------------------------
// Templated GEMM: Out[r,n] = act( sum_k A[r,k]*W[k,n] + bias[n]*bias_scale
//                                 (+ Out[r,n] if ACC) )
// 128-thread workgroup = 4 waves, each wave owns one 16-row tile and computes
// a 16x64 output strip: the A tile is loaded once per k-step and reused by
// 4 back-to-back v_wmma_f32_16x16x32_bf16 against 4 B column tiles.
// W is [K, NOUT] row-major.
// ---------------------------------------------------------------------------
template <int K, int LDA, int NOUT, int LDO, bool RELU, bool ACC, bool HASBIAS>
__global__ __launch_bounds__(128) void gemm_bf16_wmma(
    const float* __restrict__ A, const float* __restrict__ W,
    const float* __restrict__ bias, float bias_scale,
    float* __restrict__ Out)
{
  constexpr int CT = 4;                    // column tiles per wave
  const int l    = threadIdx.x & 31;
  const int wave = threadIdx.x >> 5;
  const int r    = l & 15;       // row within A tile == col within B/C tile
  const int half = l >> 4;       // lane half selects K-range / M-range
  const int row0 = (blockIdx.x * 4 + wave) * 16;
  const int col0 = blockIdx.y * (16 * CT);

  v8f c[CT];
#pragma unroll
  for (int t = 0; t < CT; ++t) {
    float bv = HASBIAS ? bias[col0 + 16 * t + r] * bias_scale : 0.0f;
#pragma unroll
    for (int j = 0; j < 8; ++j) c[t][j] = bv;
  }

#pragma unroll
  for (int kk = 0; kk < K; kk += 32) {
    // --- A tile 16x32 bf16 (ISA 7.12.2): lane r holds row r; lanes 0-15
    //     carry K={0..7,16..23}, lanes 16-31 K={8..15,24..31}. Two float4
    //     loads cover each lane's contiguous 8-float K-groups. ---
    const float4* Ap4 =
        (const float4*)(A + (size_t)(row0 + r) * LDA + kk) + half * 2;
    float4 a0 = Ap4[0];       // K = half*8 + {0..3}
    float4 a1 = Ap4[1];       // K = half*8 + {4..7}
    float4 a2 = Ap4[4];       // K = 16 + half*8 + {0..3}
    float4 a3 = Ap4[5];       // K = 16 + half*8 + {4..7}
    v16bf a;
    a[0]  = (__bf16)a0.x; a[1]  = (__bf16)a0.y;
    a[2]  = (__bf16)a0.z; a[3]  = (__bf16)a0.w;
    a[4]  = (__bf16)a1.x; a[5]  = (__bf16)a1.y;
    a[6]  = (__bf16)a1.z; a[7]  = (__bf16)a1.w;
    a[8]  = (__bf16)a2.x; a[9]  = (__bf16)a2.y;
    a[10] = (__bf16)a2.z; a[11] = (__bf16)a2.w;
    a[12] = (__bf16)a3.x; a[13] = (__bf16)a3.y;
    a[14] = (__bf16)a3.z; a[15] = (__bf16)a3.w;

    // --- B tiles 32x16 bf16: lane col = l&15; lanes 0-15 K={0..7,16..23},
    //     lanes 16-31 K={8..15,24..31}, 2 K packed per VGPR. A is reused 4x. ---
#pragma unroll
    for (int t = 0; t < CT; ++t) {
      const float* Wp =
          W + (size_t)(kk + half * 8) * NOUT + col0 + 16 * t + r;
      v16bf b;
#pragma unroll
      for (int j = 0; j < 8; ++j) {
        b[2 * j]     = (__bf16)Wp[(size_t)(2 * j + (j >= 4 ? 8 : 0)) * NOUT];
        b[2 * j + 1] = (__bf16)Wp[(size_t)(2 * j + 1 + (j >= 4 ? 8 : 0)) * NOUT];
      }
      c[t] = __builtin_amdgcn_wmma_f32_16x16x32_bf16(false, a, false, b,
                                                     (short)0, c[t], false,
                                                     false);
    }
  }

  // --- C/D layout: VGPR j -> row (j + 8*half), col (l&15) ---
  float* Op = Out + (size_t)(row0 + 8 * half) * LDO + col0 + r;
#pragma unroll
  for (int t = 0; t < CT; ++t) {
#pragma unroll
    for (int j = 0; j < 8; ++j) {
      float v = c[t][j];
      if (ACC)  v += Op[(size_t)j * LDO + 16 * t];
      if (RELU) v = fmaxf(v, 0.0f);
      Op[(size_t)j * LDO + 16 * t] = v;
    }
  }
}

// ---------------------------------------------------------------------------
// Async global->LDS staging (CDNA5): b128 per op, tracked by ASYNCcnt.
// Builtin signature (from toolchain diagnostic): param0 = AS1 (global) v4i*,
// param1 = AS3 (LDS) v4i*, then imm offset, imm cpol.
// ---------------------------------------------------------------------------
#if __has_builtin(__builtin_amdgcn_global_load_async_to_lds_b128)
#define ASYNC_CP(ldsOff, gptr)                                              \
  __builtin_amdgcn_global_load_async_to_lds_b128(                           \
      (__attribute__((address_space(1))) v4i*)(uintptr_t)(gptr),            \
      (__attribute__((address_space(3))) v4i*)(uintptr_t)(ldsOff), 0, 0)
#else
#define ASYNC_CP(ldsOff, gptr)                                              \
  asm volatile("global_load_async_to_lds_b128 %0, %1, off" ::               \
                   "v"((uint32_t)(ldsOff)), "v"(gptr) : "memory")
#endif

// ---------------------------------------------------------------------------
// Pairwise relation sum: S[b,n,h] = sum_m relu(XL[b,n,h] + YL[b,m,h])
// (rb1 folded into XL). Double-buffered async LDS staging of 32x128 y-chunks;
// each thread keeps 16 n-accumulators -> 32 VALU ops per LDS broadcast.
// ---------------------------------------------------------------------------
__global__ __launch_bounds__(256) void pairwise_sum(
    const float* __restrict__ XL, const float* __restrict__ YL,
    float* __restrict__ S)
{
  const int b  = blockIdx.y;
  const int n0 = blockIdx.x * 32;
  const int h  = threadIdx.x & (HH - 1);
  const int ns = threadIdx.x >> 7;   // 0 or 1
  __shared__ float ych[2][32 * HH];  // 2 x 16KB

  float a[16], acc[16];
#pragma unroll
  for (int i = 0; i < 16; ++i) {
    int n  = n0 + ns + 2 * i;
    a[i]   = XL[((size_t)b * NN + n) * HH + h];
    acc[i] = 0.0f;
  }

  const float* ybase = YL + (size_t)b * NN * HH;
  const uint32_t lds0 = (uint32_t)(uintptr_t)&ych[0][0];

  // issue one 32x128 chunk (4096 floats = 4 b128 ops per thread)
  auto issue = [&](int chunk) {
    uint32_t dst = lds0 + (uint32_t)(chunk & 1) * (32 * HH * 4);
#pragma unroll
    for (int t = 0; t < 4; ++t) {
      uint32_t fidx = (threadIdx.x + t * 256) * 4;           // float index
      ASYNC_CP(dst + fidx * 4, ybase + (size_t)chunk * 32 * HH + fidx);
    }
  };

  issue(0);
  for (int chunk = 0; chunk < NN / 32; ++chunk) {
    if (chunk + 1 < NN / 32) {
      issue(chunk + 1);                       // prefetch next chunk
      asm volatile("s_wait_asynccnt 4" ::: "memory");   // chunk landed
    } else {
      asm volatile("s_wait_asynccnt 0" ::: "memory");
    }
    __syncthreads();                          // all waves' data visible
    const float* yc = &ych[chunk & 1][0];
#pragma unroll
    for (int mm = 0; mm < 32; ++mm) {
      float y = yc[mm * HH + h];
#pragma unroll
      for (int i = 0; i < 16; ++i) acc[i] += fmaxf(a[i] + y, 0.0f);
    }
    __syncthreads();                          // done reading before overwrite
  }

#pragma unroll
  for (int i = 0; i < 16; ++i) {
    int n = n0 + ns + 2 * i;
    S[((size_t)b * NN + n) * HH + h] = acc[i];
  }
}

// ---------------------------------------------------------------------------
// Final: pooled[b,l] = sum_n Z[b,n,l]; out[b] = pooled . dec_w + dec_b
// ---------------------------------------------------------------------------
__global__ __launch_bounds__(64) void pool_decode(
    const float* __restrict__ Z, const float* __restrict__ dec_w,
    const float* __restrict__ dec_b, float* __restrict__ out)
{
  int b = blockIdx.x;
  int l = threadIdx.x;   // 64 threads == L
  float s = 0.0f;
  for (int n = 0; n < NN; ++n) s += Z[((size_t)b * NN + n) * LL + l];
  s *= dec_w[l];
  __shared__ float red[64];
  red[l] = s;
  __syncthreads();
  for (int off = 32; off > 0; off >>= 1) {
    if (l < off) red[l] += red[l + off];
    __syncthreads();
  }
  if (l == 0) out[b] = red[0] + dec_b[0];
}

// ---------------------------------------------------------------------------
extern "C" void kernel_launch(void* const* d_in, const int* in_sizes, int n_in,
                              void* d_out, int out_size, void* d_ws, size_t ws_size,
                              hipStream_t stream)
{
  const float* X      = (const float*)d_in[0];
  const float* proj_w = (const float*)d_in[1];
  const float* proj_b = (const float*)d_in[2];
  const float* rn_w1  = (const float*)d_in[3];   // [2, 128, 128]
  const float* rn_b1  = (const float*)d_in[4];   // [2, 128]
  const float* rn_w2  = (const float*)d_in[5];   // [2, 128, 64]
  const float* rn_b2  = (const float*)d_in[6];   // [2, 64]
  const float* fc_w1  = (const float*)d_in[7];   // [2, 64, 128]
  const float* fc_b1  = (const float*)d_in[8];   // [2, 128]
  const float* fc_w2  = (const float*)d_in[9];   // [2, 128, 64]
  const float* fc_b2  = (const float*)d_in[10];  // [2, 64]
  const float* dec_w  = (const float*)d_in[11];
  const float* dec_b  = (const float*)d_in[12];
  float* out = (float*)d_out;

  const int R = BB * NN;                 // 2048 flattened rows
  float* Z  = (float*)d_ws;              // [2048, 64]
  float* XL = Z  + (size_t)R * LL;       // [2048, 128]
  float* YL = XL + (size_t)R * HH;       // [2048, 128]
  float* S  = YL + (size_t)R * HH;       // [2048, 128]
  float* T  = S  + (size_t)R * HH;       // [2048, 128]

  dim3 wblk(128);                        // 4 row-tile waves per workgroup
  dim3 gN64(R / 64, 1);                  // Nout = 64  (one 64-col strip)
  dim3 gN128(R / 64, 2);                 // Nout = 128 (two 64-col strips)

  // Z = X @ proj_w + proj_b
  gemm_bf16_wmma<DIN, DIN, LL, LL, false, false, true>
      <<<gN64, wblk, 0, stream>>>(X, proj_w, proj_b, 1.0f, Z);

  for (int i = 0; i < 2; ++i) {
    const float* rw1 = rn_w1 + (size_t)i * 2 * LL * HH;
    // y_lin = Z @ rw1[:L]
    gemm_bf16_wmma<LL, LL, HH, HH, false, false, false>
        <<<gN128, wblk, 0, stream>>>(Z, rw1, nullptr, 0.0f, YL);
    // x_lin = Z @ rw1[L:] + rb1
    gemm_bf16_wmma<LL, LL, HH, HH, false, false, true>
        <<<gN128, wblk, 0, stream>>>(Z, rw1 + (size_t)LL * HH,
                                     rn_b1 + (size_t)i * HH, 1.0f, XL);
    // S = sum_m relu(x_lin + y_lin)
    pairwise_sum<<<dim3(NN / 32, BB), dim3(256), 0, stream>>>(XL, YL, S);
    // Z += S @ rw2 + 512 * rb2
    gemm_bf16_wmma<HH, HH, LL, LL, false, true, true>
        <<<gN64, wblk, 0, stream>>>(S, rn_w2 + (size_t)i * HH * LL,
                                    rn_b2 + (size_t)i * LL, (float)NN, Z);
    // T = relu(Z @ fw1 + fb1)
    gemm_bf16_wmma<LL, LL, HH, HH, true, false, true>
        <<<gN128, wblk, 0, stream>>>(Z, fc_w1 + (size_t)i * LL * HH,
                                     fc_b1 + (size_t)i * HH, 1.0f, T);
    // Z += T @ fw2 + fb2
    gemm_bf16_wmma<HH, HH, LL, LL, false, true, true>
        <<<gN64, wblk, 0, stream>>>(T, fc_w2 + (size_t)i * HH * LL,
                                    fc_b2 + (size_t)i * LL, 1.0f, Z);
  }

  // out[b] = (sum_n Z[b,n,:]) . dec_w + dec_b
  pool_decode<<<dim3(BB), dim3(64), 0, stream>>>(Z, dec_w, dec_b, out);
}
